// Block_18657337934741
// MI455X (gfx1250) — matrix operand order repaired and checked by reading, not compiled, not run on hardware
//
#include <hip/hip_runtime.h>

// ---------------------------------------------------------------------------
// Types for CDNA5 WMMA (gfx1250, wave32)
// ---------------------------------------------------------------------------
typedef __attribute__((ext_vector_type(16))) __bf16       bf16x16;
typedef __attribute__((ext_vector_type(8)))  float        f32x8;
typedef __attribute__((ext_vector_type(4)))  unsigned int u32x4;

union Frag { bf16x16 v; u32x4 u[2]; };

__device__ __forceinline__ unsigned short f2bf(float f) {
  unsigned int u = __builtin_bit_cast(unsigned int, f);
  u += 0x7FFFu + ((u >> 16) & 1u);           // round-to-nearest-even
  return (unsigned short)(u >> 16);
}

__device__ __forceinline__ f32x8 wmma_bf16(const Frag& a, const Frag& b, f32x8 c) {
  return __builtin_amdgcn_wmma_f32_16x16x32_bf16(
      false, a.v, false, b.v, (short)0, c, false, false);
}

// --- CDNA5 async global->LDS copy (ASYNCcnt path, ISA 10.4 / 15.18.3) ------
__device__ __forceinline__ void async_b128(unsigned lds_addr, unsigned long long gaddr) {
  asm volatile("global_load_async_to_lds_b128 %0, %1, off"
               :: "v"(lds_addr), "v"(gaddr) : "memory");
}
__device__ __forceinline__ void wait_async0() {
  asm volatile("s_wait_asynccnt 0" ::: "memory");
}
__device__ __forceinline__ unsigned lds_addr_of(const void* p) {
  return (unsigned)(unsigned long long)p;   // LDS offset = low 32 bits of flat addr
}

#define F32X8_ZERO {0.f,0.f,0.f,0.f,0.f,0.f,0.f,0.f}

// Problem constants
#define BB 2
#define TT 2048
#define DD 1024
#define HH 16
#define HD 64
#define MM (BB*TT)          // 4096 rows

// ---------------------------------------------------------------------------
// 1) Time-embedding MLP (tiny)
// ---------------------------------------------------------------------------
__global__ void time_fc1_kernel(const float* __restrict__ te, const float* __restrict__ w1,
                                const float* __restrict__ b1, float* __restrict__ u) {
  int idx = blockIdx.x * blockDim.x + threadIdx.x;       // 0..2047
  int b = idx >> 10, j = idx & 1023;
  const float* e = te + b * 1024;
  float a = b1[j], g = b1[j + 1024];
  for (int k = 0; k < 1024; ++k) {
    float ev = e[k];
    a += ev * w1[k * 2048 + j];
    g += ev * w1[k * 2048 + j + 1024];
  }
  u[b * 1024 + j] = a * g / (1.f + __expf(-g));          // a * silu(g)
}

__global__ void time_fc2_kernel(const float* __restrict__ u, const float* __restrict__ w2,
                                const float* __restrict__ b2, float* __restrict__ tp) {
  int idx = blockIdx.x * blockDim.x + threadIdx.x;       // 0..8191
  int b = idx >> 12, n = idx & 4095;
  const float* ur = u + b * 1024;
  float acc = b2[n];
  for (int k = 0; k < 1024; ++k) acc += ur[k] * w2[k * 4096 + n];
  tp[b * 4096 + n] = acc;
}

// ---------------------------------------------------------------------------
// 2) fp32 (K x N) -> bf16 transposed (N x K), LDS-tiled, coalesced both sides.
// ---------------------------------------------------------------------------
__global__ __launch_bounds__(256) void transpose_cvt_kernel(
    const float* __restrict__ in, unsigned short* __restrict__ out, int K, int N) {
  __shared__ float tile[32][33];
  int n0 = blockIdx.x * 32, k0 = blockIdx.y * 32;
  int tx = threadIdx.x & 31, ty = threadIdx.x >> 5;      // ty: 0..7
  #pragma unroll
  for (int i = 0; i < 4; ++i)
    tile[ty + 8 * i][tx] = in[(size_t)(k0 + ty + 8 * i) * N + n0 + tx];
  __syncthreads();
  #pragma unroll
  for (int i = 0; i < 4; ++i)
    out[(size_t)(n0 + ty + 8 * i) * K + k0 + tx] = f2bf(tile[tx][ty + 8 * i]);
}

// ---------------------------------------------------------------------------
// 3) RMSNorm + adaLN modulate
// ---------------------------------------------------------------------------
__global__ void modnorm_kernel(const float* __restrict__ x, const float* __restrict__ g,
                               const float* __restrict__ tp, int shift_off, int scale_off,
                               unsigned short* __restrict__ out) {
  int row = blockIdx.x;                 // 0..4095
  int b = row >> 11;
  const float* xr = x + (size_t)row * DD;
  int tid = threadIdx.x;
  float v[4]; float ss = 0.f;
  #pragma unroll
  for (int i = 0; i < 4; ++i) { v[i] = xr[tid + i * 256]; ss += v[i] * v[i]; }
  #pragma unroll
  for (int off = 16; off; off >>= 1) ss += __shfl_xor(ss, off, 32);
  __shared__ float red[8];
  int wave = tid >> 5, lane = tid & 31;
  if (lane == 0) red[wave] = ss;
  __syncthreads();
  float tot = 0.f;
  #pragma unroll
  for (int w = 0; w < 8; ++w) tot += red[w];
  float rs = rsqrtf(tot * (1.0f / 1024.0f) + 1.1920929e-07f);
  const float* shift = tp + b * 4096 + shift_off;
  const float* scale = tp + b * 4096 + scale_off;
  unsigned short* orow = out + (size_t)row * DD;
  #pragma unroll
  for (int i = 0; i < 4; ++i) {
    int j = tid + i * 256;
    float val = v[i] * rs * g[j] * (1.0f + scale[j]) + shift[j];
    orow[j] = f2bf(val);
  }
}

// ---------------------------------------------------------------------------
// 4) WMMA GEMM:  C[M,N] = A[M,K](bf16) @ Wt[N,K]^T (bf16, pre-transposed) + b
//    EPI 0: f32   EPI 1: f32 + residual   EPI 3: SwiGLU (dual tile) -> bf16
//    128x128 block tile, BK=32, 8 waves (wave tile 32x64 = 2x4 WMMA tiles),
//    double-buffered LDS fed by global_load_async_to_lds_b128.
// ---------------------------------------------------------------------------
#define BM 128
#define BN 128
#define BK 32
#define LDT_S 40          // padded LDS tile row stride (bf16 units)

template<int EPI>
__global__ __launch_bounds__(256) void gemm_wmma_kernel(
    const unsigned short* __restrict__ A, const unsigned short* __restrict__ Wt,
    const float* __restrict__ bias, const float* __restrict__ resid,
    float* __restrict__ Cf, unsigned short* __restrict__ Cb,
    int N, int K, int gate_off) {
  __shared__ __align__(16) unsigned short lds_a[2][BM * LDT_S];
  __shared__ __align__(16) unsigned short lds_b[2][BN * LDT_S];
  __shared__ __align__(16) unsigned short lds_bg[(EPI == 3) ? 2 : 1][(EPI == 3) ? BN * LDT_S : 16];

  const int tid  = threadIdx.x;
  const int lane = tid & 31;
  const int wave = tid >> 5;
  const int half = lane >> 4, l16 = lane & 15;
  const int wm = (wave & 3) * 32;
  const int wn = (wave >> 2) * 64;
  const int n0 = blockIdx.x * BN;
  const int m0 = blockIdx.y * BM;

  const f32x8 fz = F32X8_ZERO;
  f32x8 acc[2][4], accg[2][4];
  #pragma unroll
  for (int mi = 0; mi < 2; ++mi)
    #pragma unroll
    for (int ni = 0; ni < 4; ++ni) { acc[mi][ni] = fz; if constexpr (EPI == 3) accg[mi][ni] = fz; }

  // tile-copy mapping: thread t copies 16 bf16 (32B) of one row
  const int c_row = tid >> 1, c_k = (tid & 1) * 16;
  const unsigned short* gA = A  + (size_t)(m0 + c_row) * K + c_k;
  const unsigned short* gB = Wt + (size_t)(n0 + c_row) * K + c_k;
  const unsigned short* gG = Wt + (size_t)(n0 + gate_off + c_row) * K + c_k;

  auto issue_tile = [&](int k0, int buf) {
    unsigned la = lds_addr_of(&lds_a[buf][c_row * LDT_S + c_k]);
    unsigned long long ga = (unsigned long long)(gA + k0);
    async_b128(la, ga); async_b128(la + 16, ga + 16);
    unsigned lb = lds_addr_of(&lds_b[buf][c_row * LDT_S + c_k]);
    unsigned long long gb = (unsigned long long)(gB + k0);
    async_b128(lb, gb); async_b128(lb + 16, gb + 16);
    if constexpr (EPI == 3) {
      unsigned lg = lds_addr_of(&lds_bg[buf][c_row * LDT_S + c_k]);
      unsigned long long gg = (unsigned long long)(gG + k0);
      async_b128(lg, gg); async_b128(lg + 16, gg + 16);
    }
  };

  issue_tile(0, 0);
  const int NT = K / BK;
  int buf = 0;
  for (int kt = 0; kt < NT; ++kt) {
    wait_async0();
    __syncthreads();                       // buf tile visible to all waves
    if (kt + 1 < NT) issue_tile((kt + 1) * BK, buf ^ 1);

    Frag af[2];
    #pragma unroll
    for (int mi = 0; mi < 2; ++mi) {
      int m = wm + mi * 16 + l16;
      af[mi].u[0] = *reinterpret_cast<const u32x4*>(&lds_a[buf][m * LDT_S + half * 16]);
      af[mi].u[1] = *reinterpret_cast<const u32x4*>(&lds_a[buf][m * LDT_S + half * 16 + 8]);
    }
    #pragma unroll
    for (int ni = 0; ni < 4; ++ni) {
      int n = wn + ni * 16 + l16;
      Frag bfm;
      bfm.u[0] = *reinterpret_cast<const u32x4*>(&lds_b[buf][n * LDT_S + half * 16]);
      bfm.u[1] = *reinterpret_cast<const u32x4*>(&lds_b[buf][n * LDT_S + half * 16 + 8]);
      #pragma unroll
      for (int mi = 0; mi < 2; ++mi) acc[mi][ni] = wmma_bf16(af[mi], bfm, acc[mi][ni]);
      if constexpr (EPI == 3) {
        Frag bfg;
        bfg.u[0] = *reinterpret_cast<const u32x4*>(&lds_bg[buf][n * LDT_S + half * 16]);
        bfg.u[1] = *reinterpret_cast<const u32x4*>(&lds_bg[buf][n * LDT_S + half * 16 + 8]);
        #pragma unroll
        for (int mi = 0; mi < 2; ++mi) accg[mi][ni] = wmma_bf16(af[mi], bfg, accg[mi][ni]);
      }
    }
    __syncthreads();                       // all waves done with buf
    buf ^= 1;
  }

  #pragma unroll
  for (int mi = 0; mi < 2; ++mi)
    #pragma unroll
    for (int ni = 0; ni < 4; ++ni) {
      int n = n0 + wn + ni * 16 + l16;
      float bs = bias[n];
      #pragma unroll
      for (int v = 0; v < 8; ++v) {
        int m = m0 + wm + mi * 16 + half * 8 + v;
        size_t off = (size_t)m * N + n;
        float val = acc[mi][ni][v] + bs;
        if constexpr (EPI == 0) {
          Cf[off] = val;
        } else if constexpr (EPI == 1) {
          Cf[off] = resid[off] + val;
        } else {
          float gg = accg[mi][ni][v] + bias[n + gate_off];
          float u = val * gg / (1.f + __expf(-gg));
          Cb[off] = f2bf(u);
        }
      }
    }
}

// ---------------------------------------------------------------------------
// 5) RoPE + repack qkv(f32, B,T,3D) -> q,k (B,H,T,64) bf16, v^T (B,H,64,T) bf16
//    NOTE: q is pre-scaled by 1/sqrt(hd) = 0.125 here, so the attention
//    S = Q*K^T WMMA output is already softmax-scaled (kills the WMMA->VALU
//    hazard NOPs + v_pk_mul that scaling in-loop cost).
// ---------------------------------------------------------------------------
__global__ void rope_pack_kernel(const float* __restrict__ qkv,
                                 unsigned short* __restrict__ qr,
                                 unsigned short* __restrict__ kr,
                                 unsigned short* __restrict__ vt) {
  int idx = blockIdx.x * blockDim.x + threadIdx.x;   // B*T*H*32
  int i = idx & 31;
  int h = (idx >> 5) & 15;
  int t = (idx >> 9) & 2047;
  int b = idx >> 20;
  const float* base = qkv + ((size_t)(b * TT + t)) * (3 * DD) + h * HD;
  float ang = (float)t * __expf(-(2.f * (float)i / 64.f) * 9.210340372f);
  float c = __cosf(ang), s = __sinf(ang);
  float q0 = base[2 * i], q1 = base[2 * i + 1];
  size_t qo = (((size_t)(b * HH + h)) * TT + t) * HD + 2 * i;
  qr[qo]     = f2bf((q0 * c - q1 * s) * 0.125f);
  qr[qo + 1] = f2bf((q1 * c + q0 * s) * 0.125f);
  float k0 = base[DD + 2 * i], k1 = base[DD + 2 * i + 1];
  kr[qo]     = f2bf(k0 * c - k1 * s);
  kr[qo + 1] = f2bf(k1 * c + k0 * s);
  float v0 = base[2 * DD + 2 * i], v1 = base[2 * DD + 2 * i + 1];
  size_t vo = (((size_t)(b * HH + h)) * HD + 2 * i) * TT + t;
  vt[vo]      = f2bf(v0);
  vt[vo + TT] = f2bf(v1);
}

// ---------------------------------------------------------------------------
// 6) Flash attention. Grid (T/128, B*H), 8 waves; wave owns 16 query rows.
//    K/V key-tiles double-buffered in LDS via async loads.
// ---------------------------------------------------------------------------
#define LDK_S 72
#define LDV_S 136
#define LDP_S 136

__global__ __launch_bounds__(256) void attn_kernel(
    const unsigned short* __restrict__ Q, const unsigned short* __restrict__ Kr,
    const unsigned short* __restrict__ Vt, unsigned short* __restrict__ O) {
  const int bh = blockIdx.y;
  const int b = bh >> 4, h = bh & 15;
  const int q0 = blockIdx.x * 128;
  const int tid = threadIdx.x, lane = tid & 31, wave = tid >> 5;
  const int half = lane >> 4, l16 = lane & 15;

  __shared__ __align__(16) unsigned short lds_k[2][128 * LDK_S];
  __shared__ __align__(16) unsigned short lds_v[2][64 * LDV_S];
  __shared__ __align__(16) unsigned short lds_p[8 * 16 * LDP_S];

  const size_t bhT = (size_t)bh * TT;

  Frag qf[2];
  {
    const unsigned short* qb = Q + (bhT + q0 + wave * 16 + l16) * HD;
    #pragma unroll
    for (int s = 0; s < 2; ++s) {
      qf[s].u[0] = *reinterpret_cast<const u32x4*>(qb + s * 32 + half * 16);
      qf[s].u[1] = *reinterpret_cast<const u32x4*>(qb + s * 32 + half * 16 + 8);
    }
  }

  const f32x8 fz = F32X8_ZERO;
  f32x8 of[4];
  #pragma unroll
  for (int nt = 0; nt < 4; ++nt) of[nt] = fz;
  float mstate[8], lstate[8];
  #pragma unroll
  for (int v = 0; v < 8; ++v) { mstate[v] = -1e30f; lstate[v] = 0.f; }

  const int krow = tid >> 1, kpart = (tid & 1) * 32;   // K tile 128x64
  const int vrow = tid >> 2, vpart = (tid & 3) * 32;   // V tile 64x128
  unsigned short* pw = &lds_p[wave * 16 * LDP_S];

  auto issue_kv = [&](int kt, int buf) {
    unsigned lk = lds_addr_of(&lds_k[buf][krow * LDK_S + kpart]);
    unsigned long long gk =
        (unsigned long long)(Kr + (bhT + kt * 128 + krow) * HD + kpart);
    #pragma unroll
    for (int c = 0; c < 4; ++c) async_b128(lk + 16 * c, gk + 16 * c);
    unsigned lv = lds_addr_of(&lds_v[buf][vrow * LDV_S + vpart]);
    unsigned long long gv =
        (unsigned long long)(Vt + ((size_t)bh * HD + vrow) * TT + kt * 128 + vpart);
    #pragma unroll
    for (int c = 0; c < 4; ++c) async_b128(lv + 16 * c, gv + 16 * c);
  };

  issue_kv(0, 0);
  int buf = 0;
  for (int kt = 0; kt < 16; ++kt) {
    wait_async0();
    __syncthreads();
    if (kt + 1 < 16) issue_kv(kt + 1, buf ^ 1);

    // S = Q * K^T  (Q pre-scaled by 1/sqrt(hd))
    f32x8 sacc[8];
    #pragma unroll
    for (int j = 0; j < 8; ++j) sacc[j] = fz;
    #pragma unroll
    for (int s = 0; s < 2; ++s)
      #pragma unroll
      for (int j = 0; j < 8; ++j) {
        Frag kf;
        const unsigned short* pk = &lds_k[buf][(j * 16 + l16) * LDK_S + s * 32 + half * 16];
        kf.u[0] = *reinterpret_cast<const u32x4*>(pk);
        kf.u[1] = *reinterpret_cast<const u32x4*>(pk + 8);
        sacc[j] = wmma_bf16(qf[s], kf, sacc[j]);
      }

    // online softmax
    float alpha[8];
    #pragma unroll
    for (int v = 0; v < 8; ++v) {
      float mx = sacc[0][v];
      #pragma unroll
      for (int j = 1; j < 8; ++j) mx = fmaxf(mx, sacc[j][v]);
      #pragma unroll
      for (int off = 1; off < 16; off <<= 1) mx = fmaxf(mx, __shfl_xor(mx, off, 32));
      float mnew = fmaxf(mstate[v], mx);
      alpha[v] = __expf(mstate[v] - mnew);
      mstate[v] = mnew;
    }
    #pragma unroll
    for (int v = 0; v < 8; ++v) {
      float rs = 0.f;
      #pragma unroll
      for (int j = 0; j < 8; ++j) {
        float p = __expf(sacc[j][v] - mstate[v]);
        rs += p;
        pw[(half * 8 + v) * LDP_S + j * 16 + l16] = f2bf(p);
      }
      #pragma unroll
      for (int off = 1; off < 16; off <<= 1) rs += __shfl_xor(rs, off, 32);
      lstate[v] = lstate[v] * alpha[v] + rs;
    }
    #pragma unroll
    for (int nt = 0; nt < 4; ++nt)
      #pragma unroll
      for (int v = 0; v < 8; ++v) of[nt][v] *= alpha[v];

    // O += P * V
    #pragma unroll
    for (int s2 = 0; s2 < 4; ++s2) {
      Frag pf;
      pf.u[0] = *reinterpret_cast<const u32x4*>(&pw[l16 * LDP_S + s2 * 32 + half * 16]);
      pf.u[1] = *reinterpret_cast<const u32x4*>(&pw[l16 * LDP_S + s2 * 32 + half * 16 + 8]);
      #pragma unroll
      for (int nt = 0; nt < 4; ++nt) {
        Frag vf;
        const unsigned short* pv = &lds_v[buf][(nt * 16 + l16) * LDV_S + s2 * 32 + half * 16];
        vf.u[0] = *reinterpret_cast<const u32x4*>(pv);
        vf.u[1] = *reinterpret_cast<const u32x4*>(pv + 8);
        of[nt] = wmma_bf16(pf, vf, of[nt]);
      }
    }
    __syncthreads();
    buf ^= 1;
  }

  #pragma unroll
  for (int v = 0; v < 8; ++v) lstate[v] = 1.f / lstate[v];
  #pragma unroll
  for (int nt = 0; nt < 4; ++nt)
    #pragma unroll
    for (int v = 0; v < 8; ++v) {
      int m = q0 + wave * 16 + half * 8 + v;
      int col = h * HD + nt * 16 + l16;
      O[((size_t)b * TT + m) * DD + col] = f2bf(of[nt][v] * lstate[v]);
    }
}

// ---------------------------------------------------------------------------
// Launch
// ---------------------------------------------------------------------------
extern "C" void kernel_launch(void* const* d_in, const int* in_sizes, int n_in,
                              void* d_out, int out_size, void* d_ws, size_t ws_size,
                              hipStream_t stream) {
  const float* x    = (const float*)d_in[0];
  const float* te   = (const float*)d_in[1];
  const float* g1   = (const float*)d_in[2];
  const float* g2   = (const float*)d_in[3];
  const float* wqkv = (const float*)d_in[4];
  const float* bqkv = (const float*)d_in[5];
  const float* wao  = (const float*)d_in[6];
  const float* bao  = (const float*)d_in[7];
  const float* wfc  = (const float*)d_in[8];
  const float* bfc  = (const float*)d_in[9];
  const float* wfo  = (const float*)d_in[10];
  const float* bfo  = (const float*)d_in[11];
  const float* wt1  = (const float*)d_in[12];
  const float* bt1  = (const float*)d_in[13];
  const float* wt2  = (const float*)d_in[14];
  const float* bt2  = (const float*)d_in[15];
  float* out = (float*)d_out;

  char* ws = (char*)d_ws;
  size_t off_tp   = 0;                          // B*4D f32
  size_t off_ut   = off_tp   + 32768;
  size_t off_hbf  = off_ut   + 8192;            // M*D bf16 (h1 then h2)
  size_t off_wqkv = off_hbf  + 8388608;         // W^T bf16: 3072 x 1024
  size_t off_wao  = off_wqkv + 6291456;         // 1024 x 1024
  size_t off_wfc  = off_wao  + 2097152;         // 8192 x 1024
  size_t off_wfo  = off_wfc  + 16777216;        // 1024 x 4096
  size_t off_qr   = off_wfo  + 8388608;
  size_t off_kr   = off_qr   + 8388608;
  size_t off_vt   = off_kr   + 8388608;
  size_t off_abf  = off_vt   + 8388608;
  size_t off_big  = off_abf  + 8388608;         // 48MB: qkv f32; later x1 + u

  float*          tp    = (float*)(ws + off_tp);
  float*          ut    = (float*)(ws + off_ut);
  unsigned short* hbf   = (unsigned short*)(ws + off_hbf);
  unsigned short* wqkvb = (unsigned short*)(ws + off_wqkv);
  unsigned short* waob  = (unsigned short*)(ws + off_wao);
  unsigned short* wfcb  = (unsigned short*)(ws + off_wfc);
  unsigned short* wfob  = (unsigned short*)(ws + off_wfo);
  unsigned short* qr    = (unsigned short*)(ws + off_qr);
  unsigned short* kr    = (unsigned short*)(ws + off_kr);
  unsigned short* vt    = (unsigned short*)(ws + off_vt);
  unsigned short* abf   = (unsigned short*)(ws + off_abf);
  float*          qkvf  = (float*)(ws + off_big);
  float*          x1    = (float*)(ws + off_big);              // reuses qkv region
  unsigned short* ubf   = (unsigned short*)(ws + off_big + 16777216);

  // 1) time MLP
  time_fc1_kernel<<<8, 256, 0, stream>>>(te, wt1, bt1, ut);
  time_fc2_kernel<<<32, 256, 0, stream>>>(ut, wt2, bt2, tp);

  // 2) weights -> bf16, transposed to (N x K)
  transpose_cvt_kernel<<<dim3(3*DD/32, DD/32),   256, 0, stream>>>(wqkv, wqkvb, DD, 3*DD);
  transpose_cvt_kernel<<<dim3(DD/32,   DD/32),   256, 0, stream>>>(wao,  waob,  DD, DD);
  transpose_cvt_kernel<<<dim3(8*DD/32, DD/32),   256, 0, stream>>>(wfc,  wfcb,  DD, 8*DD);
  transpose_cvt_kernel<<<dim3(DD/32,   4*DD/32), 256, 0, stream>>>(wfo,  wfob,  4*DD, DD);

  // 3) h1 = modulate(rmsnorm(x,g1); shift1, scale1)
  modnorm_kernel<<<MM, 256, 0, stream>>>(x, g1, tp, 0, 1024, hbf);

  // 4) qkv = h1 @ Wqkv + b   (f32)
  gemm_wmma_kernel<0><<<dim3(3*DD/BN, MM/BM), 256, 0, stream>>>(
      hbf, wqkvb, bqkv, nullptr, qkvf, nullptr, 3*DD, DD, 0);

  // 5) RoPE + pack (q pre-scaled by 1/sqrt(hd))
  rope_pack_kernel<<<(BB*TT*HH*32)/256, 256, 0, stream>>>(qkvf, qr, kr, vt);

  // 6) flash attention -> abf (bf16, B,T,D)
  attn_kernel<<<dim3(TT/128, BB*HH), 256, 0, stream>>>(qr, kr, vt, abf);

  // 7) x1 = x + attn @ Wao + b
  gemm_wmma_kernel<1><<<dim3(DD/BN, MM/BM), 256, 0, stream>>>(
      abf, waob, bao, x, x1, nullptr, DD, DD, 0);

  // 8) h2 = modulate(rmsnorm(x1,g2); shift2, scale2)
  modnorm_kernel<<<MM, 256, 0, stream>>>(x1, g2, tp, 2048, 3072, hbf);

  // 9) u = swiglu(h2 @ Wfc + b)  (bf16, M x 4096)
  gemm_wmma_kernel<3><<<dim3(4*DD/BN, MM/BM), 256, 0, stream>>>(
      hbf, wfcb, bfc, nullptr, nullptr, ubf, 4*DD, DD, 4*DD);

  // 10) out = x1 + u @ Wfo + b
  gemm_wmma_kernel<1><<<dim3(DD/BN, MM/BM), 256, 0, stream>>>(
      ubf, wfob, bfo, x1, out, nullptr, DD, 4*DD, 0);
}
